// InnerSelfAttention_81707457839480
// MI455X (gfx1250) — compile-verified, matrix operand order checked by reading, not run on hardware
//
#include <hip/hip_runtime.h>

// ---------------------------------------------------------------------------
// MI455X (gfx1250) sliding-window attention, compute-bound (~39 GFLOP vs
// ~50 MB traffic -> push everything through v_wmma_f32_16x16x32_bf16).
// Pipeline: cast -> Q/K/V WMMA GEMMs (V stored transposed [B,H,D,S]) ->
// flash attention (wave32, per-16-query-tile, LDS reshape of P) ->
// output-projection WMMA GEMM (+bias, fp32 out).
//
// GEMM: wave tile 16x64 (1 A-frag reused by 4 B-frags -> 4 back-to-back
// WMMAs), 2x-unrolled ping-pong double buffering: no register-rotation moves,
// WMMAs overlap the other buffer's loads (partial s_wait_loadcnt).
// ---------------------------------------------------------------------------

typedef __attribute__((ext_vector_type(16))) __bf16 v16bf;
typedef __attribute__((ext_vector_type(8)))  float  v8f;
typedef __attribute__((ext_vector_type(8)))  __bf16 bf16x8;
typedef __attribute__((ext_vector_type(4)))  __bf16 bf16x4;

#define WMMA_BF16(a, b, c) \
  __builtin_amdgcn_wmma_f32_16x16x32_bf16(false, (a), false, (b), (short)0, (c), false, false)

constexpr int kB = 2, kS = 2048, kE = 1024, kH = 16, kD = 64, kW = 256;

// Load one 16x32 bf16 WMMA operand fragment.
// ISA layout: elements 0..7  = K[khalf*8 .. khalf*8+7]
//             elements 8..15 = K[16+khalf*8 .. 16+khalf*8+7]
template <typename P>
__device__ inline v16bf load_frag(P rowbase, int khalf) {
  bf16x8 lo = *(const bf16x8*)(rowbase + khalf * 8);
  bf16x8 hi = *(const bf16x8*)(rowbase + 16 + khalf * 8);
  v16bf f;
#pragma unroll
  for (int t = 0; t < 8; ++t) { f[t] = lo[t]; f[t + 8] = hi[t]; }
  return f;
}

// ---------------------------------------------------------------------------
// fp32 -> bf16 cast, 4 elements / thread
// ---------------------------------------------------------------------------
__global__ __launch_bounds__(256) void cast_f32_bf16_kernel(
    const float* __restrict__ in, __bf16* __restrict__ out, int n) {
  int i = (blockIdx.x * 256 + threadIdx.x) * 4;
  if (i + 3 < n) {
    float4 v = *(const float4*)(in + i);
    bf16x4 o;
    o[0] = (__bf16)v.x; o[1] = (__bf16)v.y; o[2] = (__bf16)v.z; o[3] = (__bf16)v.w;
    *(bf16x4*)(out + i) = o;
  }
}

// ---------------------------------------------------------------------------
// C[M,N] = A[M,K] @ W^T  (W is [N,K], nn.Linear layout) via WMMA bf16.
//   mode 0: store bf16 row-major [M,N]           (Q, K projections)
//   mode 1: store bf16 transposed per head       (V -> Vt[B,H,D,S])
//   mode 2: store fp32 + bias                    (output projection)
// Block = 256 threads = 8 waves arranged 2(M) x 4(N); wave tile = 16x64.
// grid = (N/256, M/32).  K-loop 2x-unrolled with ping-pong buffers.
// ---------------------------------------------------------------------------
__global__ __launch_bounds__(256) void gemm_wmma_kernel(
    const __bf16* __restrict__ A, const __bf16* __restrict__ W,
    __bf16* __restrict__ Cb, float* __restrict__ Cf,
    const float* __restrict__ bias, int mode) {
  constexpr int K = kE, N = kE;
  const int wave = threadIdx.x >> 5;
  const int lane = threadIdx.x & 31;
  const int lane16 = lane & 15;
  const int khalf  = lane >> 4;
  const int waveM = wave >> 2;   // 0..1
  const int waveN = wave & 3;    // 0..3

  const int m0 = blockIdx.y * 32 + waveM * 16;
  const int n0 = blockIdx.x * 256 + waveN * 64;

  const __bf16* arow = A + (size_t)(m0 + lane16) * K;  // A: row m, contiguous K
  const __bf16* wrow[4];
#pragma unroll
  for (int j = 0; j < 4; ++j)                          // B[k,n] = W[n,k]
    wrow[j] = W + (size_t)(n0 + j * 16 + lane16) * K;

  v8f acc[4] = {{}, {}, {}, {}};

  // ping-pong fragment buffers
  v16bf a0, a1, b0[4], b1[4];

  // prologue: buffer0 <- step k=0
  a0 = load_frag(arow, khalf);
#pragma unroll
  for (int j = 0; j < 4; ++j) b0[j] = load_frag(wrow[j], khalf);

  // steady state: 15 iterations covering steps 32..928 and loading 64..960
  for (int k = 32; k < K - 32; k += 64) {
    __builtin_prefetch(arow + k + 224, 0, 1);   // speculative, global_prefetch_b8
    __builtin_prefetch(wrow[0] + k + 224, 0, 1);
    __builtin_prefetch(wrow[1] + k + 224, 0, 1);
    __builtin_prefetch(wrow[2] + k + 224, 0, 1);
    __builtin_prefetch(wrow[3] + k + 224, 0, 1);

    // load buffer1 <- step k, compute buffer0 (step k-32)
    a1 = load_frag(arow + k, khalf);
#pragma unroll
    for (int j = 0; j < 4; ++j) b1[j] = load_frag(wrow[j] + k, khalf);
#pragma unroll
    for (int j = 0; j < 4; ++j) acc[j] = WMMA_BF16(a0, b0[j], acc[j]);

    // load buffer0 <- step k+32, compute buffer1 (step k)
    a0 = load_frag(arow + k + 32, khalf);
#pragma unroll
    for (int j = 0; j < 4; ++j) b0[j] = load_frag(wrow[j] + k + 32, khalf);
#pragma unroll
    for (int j = 0; j < 4; ++j) acc[j] = WMMA_BF16(a1, b1[j], acc[j]);
  }

  // epilogue: buffer0 holds step K-64; load buffer1 <- step K-32
  {
    const int k = K - 32;
    a1 = load_frag(arow + k, khalf);
#pragma unroll
    for (int j = 0; j < 4; ++j) b1[j] = load_frag(wrow[j] + k, khalf);
#pragma unroll
    for (int j = 0; j < 4; ++j) acc[j] = WMMA_BF16(a0, b0[j], acc[j]);
#pragma unroll
    for (int j = 0; j < 4; ++j) acc[j] = WMMA_BF16(a1, b1[j], acc[j]);
  }

  // D layout: col n = lane16 within each 16-wide tile; rows = r + 8*khalf
  const int mrow = m0 + 8 * khalf;
  if (mode == 0) {
#pragma unroll
    for (int nt = 0; nt < 4; ++nt) {
      const int n = n0 + nt * 16 + lane16;
#pragma unroll
      for (int r = 0; r < 8; ++r)
        Cb[(size_t)(mrow + r) * N + n] = (__bf16)acc[nt][r];
    }
  } else if (mode == 1) {
    // Vt[b, h, d, s]: lane's 8 rows are contiguous in s -> one 16B store
    const int b = mrow / kS, s = mrow % kS;
#pragma unroll
    for (int nt = 0; nt < 4; ++nt) {
      const int n = n0 + nt * 16 + lane16;
      const int h = n >> 6, d = n & 63;
      bf16x8 v;
#pragma unroll
      for (int r = 0; r < 8; ++r) v[r] = (__bf16)acc[nt][r];
      *(bf16x8*)(Cb + ((size_t)(b * kH + h) * kD + d) * kS + s) = v;
    }
  } else {
#pragma unroll
    for (int nt = 0; nt < 4; ++nt) {
      const int n = n0 + nt * 16 + lane16;
      const float bv = bias[n];
#pragma unroll
      for (int r = 0; r < 8; ++r)
        Cf[(size_t)(mrow + r) * N + n] = acc[nt][r] + bv;
    }
  }
}

// ---------------------------------------------------------------------------
// Flash attention over the causal band (j <= i && j > i-256), no 1/sqrt(d).
// One wave per (b, h, 16-query tile); 4 waves / block.
// Per 32-key chunk: 4 WMMAs for S=QK^T, fp32 online softmax (half-wave
// shuffles), P reshaped via LDS into an A-fragment, 4 WMMAs for O += P.V.
// ---------------------------------------------------------------------------
__global__ __launch_bounds__(128) void flash_attn_kernel(
    const __bf16* __restrict__ Q, const __bf16* __restrict__ Km,
    const __bf16* __restrict__ Vt, __bf16* __restrict__ Ao) {
  __shared__ __bf16 plds[4][16 * 32];  // per-wave P staging (1 KB each)

  const int wave = threadIdx.x >> 5;
  const int lane = threadIdx.x & 31;
  const int lane16 = lane & 15;
  const int khalf  = lane >> 4;

  const int QT = kS / 16;
  const int g  = blockIdx.x * 4 + wave;
  const int qt = g % QT;
  const int h  = (g / QT) % kH;
  const int b  = g / (QT * kH);
  const int q0 = qt * 16;

  // Q fragments (M=16 queries, K=64 -> two 16x32 frags)
  const __bf16* qrow = Q + (size_t)(b * kS + q0 + lane16) * kE + h * kD;
  const v16bf qa0 = load_frag(qrow, khalf);
  const v16bf qa1 = load_frag(qrow + 32, khalf);

  v8f o0 = {}, o1 = {}, o2 = {}, o3 = {};
  float mi[8], li[8];
#pragma unroll
  for (int r = 0; r < 8; ++r) { mi[r] = -1e30f; li[r] = 0.0f; }

  int c0 = q0 - (kW - 1);
  if (c0 < 0) c0 = 0;
  c0 &= ~31;

  __bf16* pl = &plds[wave][0];

  for (int c = c0; c <= q0 + 15; c += 32) {
    // --- scores: two 16x16 tiles covering keys [c, c+31] ---
    v8f sf[2];
#pragma unroll
    for (int t = 0; t < 2; ++t) {
      int ki = c + t * 16 + lane16;
      if (ki > kS - 1) ki = kS - 1;  // clamp (masked below anyway)
      const __bf16* krow = Km + (size_t)(b * kS + ki) * kE + h * kD;
      v16bf bk0 = load_frag(krow, khalf);
      v16bf bk1 = load_frag(krow + 32, khalf);
      v8f cz = {};
      cz = WMMA_BF16(qa0, bk0, cz);
      cz = WMMA_BF16(qa1, bk1, cz);
      sf[t] = cz;
    }

    // --- band mask + online softmax (rows live in 16-lane half-groups) ---
#pragma unroll
    for (int r = 0; r < 8; ++r) {
      const int i = q0 + r + 8 * khalf;
      const int j0 = c + lane16, j1 = c + 16 + lane16;
      float s0 = sf[0][r], s1 = sf[1][r];
      if (!((j0 <= i) && (j0 + kW > i))) s0 = -1e30f;
      if (!((j1 <= i) && (j1 + kW > i))) s1 = -1e30f;

      float m = fmaxf(s0, s1);
#pragma unroll
      for (int off = 1; off < 16; off <<= 1)
        m = fmaxf(m, __shfl_xor(m, off, 32));

      const float mn = fmaxf(mi[r], m);
      const float sc = __expf(mi[r] - mn);
      const float p0 = (s0 > -1e29f) ? __expf(s0 - mn) : 0.0f;
      const float p1 = (s1 > -1e29f) ? __expf(s1 - mn) : 0.0f;
      float rs = p0 + p1;
#pragma unroll
      for (int off = 1; off < 16; off <<= 1)
        rs += __shfl_xor(rs, off, 32);

      mi[r] = mn;
      li[r] = li[r] * sc + rs;
      o0[r] *= sc; o1[r] *= sc; o2[r] *= sc; o3[r] *= sc;

      // stash P tile (D-layout -> LDS row-major 16x32)
      const int row = r + 8 * khalf;
      pl[row * 32 + lane16]      = (__bf16)p0;
      pl[row * 32 + 16 + lane16] = (__bf16)p1;
    }
    asm volatile("s_wait_dscnt 0" ::: "memory");

    // --- reload P as A-fragment (16 queries x 32 keys) ---
    const v16bf pa = load_frag(pl + lane16 * 32, khalf);

    // --- O += P . V ; Vt rows are contiguous along keys ---
    const __bf16* vbase = Vt + (size_t)(b * kH + h) * kD * kS;
#pragma unroll
    for (int nt = 0; nt < 4; ++nt) {
      const int d = nt * 16 + lane16;
      const __bf16* vrow = vbase + (size_t)d * kS + c;
      v16bf bv = load_frag(vrow, khalf);
      if      (nt == 0) o0 = WMMA_BF16(pa, bv, o0);
      else if (nt == 1) o1 = WMMA_BF16(pa, bv, o1);
      else if (nt == 2) o2 = WMMA_BF16(pa, bv, o2);
      else              o3 = WMMA_BF16(pa, bv, o3);
    }
    asm volatile("s_wait_dscnt 0" ::: "memory");  // pl reads done before rewrite
  }

  // --- normalize + store merged-head bf16 [B,S,E] for the Wo GEMM ---
#pragma unroll
  for (int r = 0; r < 8; ++r) {
    const int row = q0 + r + 8 * khalf;
    const float inv = 1.0f / li[r];
    __bf16* op = Ao + (size_t)(b * kS + row) * kE + h * kD;
    op[lane16]      = (__bf16)(o0[r] * inv);
    op[16 + lane16] = (__bf16)(o1[r] * inv);
    op[32 + lane16] = (__bf16)(o2[r] * inv);
    op[48 + lane16] = (__bf16)(o3[r] * inv);
  }
}

// ---------------------------------------------------------------------------
extern "C" void kernel_launch(void* const* d_in, const int* in_sizes, int n_in,
                              void* d_out, int out_size, void* d_ws, size_t ws_size,
                              hipStream_t stream) {
  const float* hs = (const float*)d_in[0];
  const float* Wq = (const float*)d_in[1];
  const float* Wk = (const float*)d_in[2];
  const float* Wv = (const float*)d_in[3];
  const float* Wo = (const float*)d_in[4];
  const float* bo = (const float*)d_in[5];
  float* out = (float*)d_out;
  (void)in_sizes; (void)n_in; (void)out_size; (void)ws_size;

  char* ws = (char*)d_ws;
  size_t off = 0;
  auto walloc = [&](size_t bytes) -> void* {
    void* p = ws + off;
    off += (bytes + 255) & ~(size_t)255;
    return p;
  };

  __bf16* hsb = (__bf16*)walloc((size_t)kB * kS * kE * 2);
  __bf16* wqb = (__bf16*)walloc((size_t)kE * kE * 2);
  __bf16* wkb = (__bf16*)walloc((size_t)kE * kE * 2);
  __bf16* wvb = (__bf16*)walloc((size_t)kE * kE * 2);
  __bf16* wob = (__bf16*)walloc((size_t)kE * kE * 2);
  __bf16* qb  = (__bf16*)walloc((size_t)kB * kS * kE * 2);
  __bf16* kb  = (__bf16*)walloc((size_t)kB * kS * kE * 2);
  __bf16* vtb = (__bf16*)walloc((size_t)kB * kS * kE * 2 + 256);  // pad for tail reads
  __bf16* aob = (__bf16*)walloc((size_t)kB * kS * kE * 2);

  // 1) fp32 -> bf16 casts
  const int nh = kB * kS * kE;   // 4,194,304
  const int nw = kE * kE;        // 1,048,576
  cast_f32_bf16_kernel<<<nh / 1024, 256, 0, stream>>>(hs, hsb, nh);
  cast_f32_bf16_kernel<<<nw / 1024, 256, 0, stream>>>(Wq, wqb, nw);
  cast_f32_bf16_kernel<<<nw / 1024, 256, 0, stream>>>(Wk, wkb, nw);
  cast_f32_bf16_kernel<<<nw / 1024, 256, 0, stream>>>(Wv, wvb, nw);
  cast_f32_bf16_kernel<<<nw / 1024, 256, 0, stream>>>(Wo, wob, nw);

  // 2) Q/K/V projections (WMMA); V stored transposed [B,H,D,S]
  dim3 gg(kE / 256, (kB * kS) / 32);  // (4, 128)
  gemm_wmma_kernel<<<gg, 256, 0, stream>>>(hsb, wqb, qb,  nullptr, nullptr, 0);
  gemm_wmma_kernel<<<gg, 256, 0, stream>>>(hsb, wkb, kb,  nullptr, nullptr, 0);
  gemm_wmma_kernel<<<gg, 256, 0, stream>>>(hsb, wvb, vtb, nullptr, nullptr, 1);

  // 3) banded flash attention
  const int nqt = kB * kH * (kS / 16);  // 4096 wave-tiles
  flash_attn_kernel<<<nqt / 4, 128, 0, stream>>>(qb, kb, vtb, aob);

  // 4) output projection + bias, fp32 out
  gemm_wmma_kernel<<<gg, 256, 0, stream>>>(aob, wob, nullptr, out, bo, 2);
}
